// DecoderBlock_83837761618016
// MI455X (gfx1250) — compile-verified, matrix operand order checked by reading, not run on hardware
//
#include <hip/hip_runtime.h>
#include <math.h>

// ---------------------------------------------------------------------------
// CDNA5 (gfx1250) WMMA types & helpers
// ---------------------------------------------------------------------------
typedef __attribute__((ext_vector_type(16))) __bf16 v16bf;
typedef __attribute__((ext_vector_type(8)))  float  v8f;

union ABu {
  v16bf          v;
  unsigned int   u[8];
  unsigned short s[16];
  __bf16         b[16];
};

union U4u {
  uint4        q;
  unsigned int u[4];
};

// k-index of the t-th contiguous bf16 pair for a lane-half, per the CDNA5
// 16-bit A/B fragment layout (lanes 0-15: K 0..7 then 16..23; lanes 16-31:
// K 8..15 then 24..31; each VGPR packs two consecutive K values).
__device__ __forceinline__ int kpair(int t, int half) {
  int base = (t < 4) ? (2 * t) : (16 + 2 * (t - 4));
  return base + 8 * half;
}

__device__ __forceinline__ v8f wmma_bf16(v16bf a, v16bf b, v8f c) {
  return __builtin_amdgcn_wmma_f32_16x16x32_bf16(false, a, false, b,
                                                 (short)0, c, false, false);
}

#define D_  1024
#define H_  16
#define DH_ 64

// ---------------------------------------------------------------------------
// fp32 -> bf16 conversion (grid-stride)
// ---------------------------------------------------------------------------
__global__ void cvt_kernel(const float* __restrict__ in,
                           __bf16* __restrict__ out,
                           unsigned long long n) {
  unsigned long long i = (unsigned long long)blockIdx.x * blockDim.x + threadIdx.x;
  unsigned long long stride = (unsigned long long)gridDim.x * blockDim.x;
  for (; i < n; i += stride) out[i] = (__bf16)in[i];
}

// ---------------------------------------------------------------------------
// Tiled bf16 GEMM: C[Mr x N] = A[Mr x K] @ W[K x N] + bias, optional GELU.
// Block tile 256x128, 8 waves (4x2), wave tile 64x64 = 4x4 WMMA fragments.
// Double-buffered LDS, one barrier per K-step. Staging is branch-free
// (last iteration redundantly re-fetches tile 0 into the idle buffer) and a
// sched_barrier(0) fences staging from the WMMA burst so the staging
// registers die early instead of being spilled to scratch across the burst.
// ---------------------------------------------------------------------------
__global__ __launch_bounds__(256, 1) void gemm_bf16_kernel(
    const __bf16* __restrict__ A, const __bf16* __restrict__ W,
    const float* __restrict__ bias,
    float* __restrict__ Cf, __bf16* __restrict__ Cb,
    int Mr, int N, int K, int act) {
  (void)Mr;
  __shared__ __align__(16) __bf16 lA[2][256][32];        // [buf][row][k]  32 KB
  __shared__ __align__(16) unsigned int lBu[2][128][16]; // [buf][col][k/2] 16 KB

  const int tid  = threadIdx.x;
  const int lane = tid & 31;
  const int wave = tid >> 5;
  const int wm   = wave >> 1;      // 0..3
  const int wn   = wave & 1;       // 0..1
  const int half = lane >> 4;
  const int lr   = lane & 15;
  const int row0 = blockIdx.y * 256;
  const int col0 = blockIdx.x * 128;

  v8f acc[4][4];
#pragma unroll
  for (int mi = 0; mi < 4; ++mi)
#pragma unroll
    for (int ni = 0; ni < 4; ++ni)
#pragma unroll
      for (int e = 0; e < 8; ++e) acc[mi][ni][e] = 0.0f;

  // staging assignment: A -> thread owns one full 32-wide row; B -> thread
  // owns a (2 k-rows) x (8 cols) patch, repacked into dwords of (k, k+1).
  const int ar  = tid;               // 0..255  A-tile row
  const int bk2 = (tid & 15) * 2;    // 0..30   even k-row
  const int bn8 = (tid >> 4) * 8;    // 0..120  col base

  uint4        aReg[4];
  unsigned int bReg[8];

  auto stageLoad = [&](int k0) {
    const uint4* As = (const uint4*)(A + (size_t)(row0 + ar) * K + k0);
#pragma unroll
    for (int e = 0; e < 4; ++e) aReg[e] = As[e];
    U4u w0, w1;
    const __bf16* Ws = W + (size_t)(k0 + bk2) * N + col0 + bn8;
    w0.q = *(const uint4*)Ws;
    w1.q = *(const uint4*)(Ws + N);
#pragma unroll
    for (int e = 0; e < 4; ++e) {
      bReg[2 * e]     = (w0.u[e] & 0xffffu) | (w1.u[e] << 16);          // col 2e
      bReg[2 * e + 1] = (w0.u[e] >> 16) | (w1.u[e] & 0xffff0000u);      // col 2e+1
    }
  };
  auto stageStore = [&](int buf) {
#pragma unroll
    for (int e = 0; e < 4; ++e) *(uint4*)&lA[buf][ar][8 * e] = aReg[e];
#pragma unroll
    for (int e = 0; e < 8; ++e) lBu[buf][bn8 + e][bk2 >> 1] = bReg[e];
  };

  stageLoad(0);
  stageStore(0);
  __syncthreads();

  const int nt = K >> 5;
  for (int it = 0; it < nt; ++it) {
    const int cur = it & 1;
    // Branch-free staging: last iteration redundantly re-fetches tile 0 into
    // the idle buffer (never read). Keeps everything in one basic block so
    // the stores cannot be sunk past the WMMA burst.
    const int knext = ((it + 1) < nt ? (it + 1) : 0) << 5;
    const int kpf   = ((it + 2) < nt ? (it + 2) : 0) << 5;
    stageLoad(knext);
    __builtin_prefetch(A + (size_t)(row0 + ar) * K + kpf, 0, 1);   // tile+2 hint
    __builtin_prefetch(W + (size_t)(kpf + bk2) * N + col0 + bn8, 0, 1);
    stageStore(1 - cur);   // idle buffer: last read in it-1, which ended in a barrier

    // Scheduling fence: staging (global loads + LDS stores) completes here;
    // aReg/bReg are dead before the high-pressure WMMA region -> no spills.
    __builtin_amdgcn_sched_barrier(0);

    // A fragments for the current tile.
    v16bf af[4];
#pragma unroll
    for (int mi = 0; mi < 4; ++mi) {
      ABu u;
      const int r = wm * 64 + mi * 16 + lr;
#pragma unroll
      for (int t = 0; t < 8; ++t)
        u.u[t] = *(const unsigned int*)&lA[cur][r][kpair(t, half)];
      af[mi] = u.v;
    }

    // 16-WMMA burst, B fragments streamed from LDS.
#pragma unroll
    for (int ni = 0; ni < 4; ++ni) {
      ABu u;
      const int n = wn * 64 + ni * 16 + lr;
#pragma unroll
      for (int t = 0; t < 8; ++t)
        u.u[t] = lBu[cur][n][kpair(t, half) >> 1];
#pragma unroll
      for (int mi = 0; mi < 4; ++mi)
        acc[mi][ni] = wmma_bf16(af[mi], u.v, acc[mi][ni]);
    }

    __syncthreads();                         // one barrier per K-step
  }

  // epilogue: bias (+GELU) and store f32 / bf16 -------------------------------
#pragma unroll
  for (int mi = 0; mi < 4; ++mi)
#pragma unroll
    for (int ni = 0; ni < 4; ++ni) {
      const int   col = col0 + wn * 64 + ni * 16 + lr;
      const float bsv = bias ? bias[col] : 0.0f;
#pragma unroll
      for (int v = 0; v < 8; ++v) {
        const int row = row0 + wm * 64 + mi * 16 + v + 8 * half;
        float val = acc[mi][ni][v] + bsv;
        if (act == 1) val = 0.5f * val * (1.0f + erff(val * 0.70710678118f));
        if (Cf) Cf[(size_t)row * N + col] = val;
        if (Cb) Cb[(size_t)row * N + col] = (__bf16)val;
      }
    }
}

// ---------------------------------------------------------------------------
// Flash attention: one wave per (batch, head, 16-query block).
// windowed=1 -> causal local window W=64 + ALiBi; windowed=0 -> full attn.
// Q,K,V,O are bf16 [B*Lq(or Lk), D] with head h at feature offset h*64.
// ---------------------------------------------------------------------------
__global__ __launch_bounds__(128) void attn_kernel(
    const __bf16* __restrict__ Q, const __bf16* __restrict__ Km,
    const __bf16* __restrict__ Vm, __bf16* __restrict__ O,
    int Lq, int Lk, int windowed) {
  __shared__ __bf16 lP[4][16][32];  // per-wave P staging (C-layout -> A-layout)
  const int lane  = threadIdx.x & 31;
  const int wave  = threadIdx.x >> 5;
  const int half  = lane >> 4;
  const int lr    = lane & 15;
  const int qbPer = Lq >> 4;
  const int job   = blockIdx.x * 4 + wave;
  const int qblk  = job % qbPer;
  const int h     = (job / qbPer) % H_;
  const int b     = job / (qbPer * H_);
  const int q0    = qblk << 4;

  const float slope = windowed ? exp2f(-8.0f * (float)(h + 1) / (float)H_) : 0.0f;

  // Q fragments (A operand): 16 queries x 64 features
  v16bf qa[2];
  {
    const __bf16* qb = Q + ((size_t)(b * Lq + q0 + lr)) * D_ + h * DH_;
#pragma unroll
    for (int f = 0; f < 2; ++f) {
      ABu u;
#pragma unroll
      for (int t = 0; t < 8; ++t)
        u.u[t] = *(const unsigned int*)(qb + f * 32 + kpair(t, half));
      qa[f] = u.v;
    }
  }

  float mrow[8], lrow[8];
  v8f   o[4];
#pragma unroll
  for (int v = 0; v < 8; ++v) { mrow[v] = -3.0e38f; lrow[v] = 0.0f; }
#pragma unroll
  for (int f = 0; f < 4; ++f)
#pragma unroll
    for (int e = 0; e < 8; ++e) o[f][e] = 0.0f;

  const int nch = windowed ? 3 : (Lk >> 5);  // uniform across the block
  for (int ch = 0; ch < nch; ++ch) {
    const int kc = windowed ? (q0 - 64 + 32 * ch) : (32 * ch);

    // S = Q @ K^T : 16 queries x 32 keys ------------------------------------
    v8f s[2];
#pragma unroll
    for (int g = 0; g < 2; ++g) {
#pragma unroll
      for (int e = 0; e < 8; ++e) s[g][e] = 0.0f;
      int key  = kc + g * 16 + lr;
      int keyc = key < 0 ? 0 : (key >= Lk ? Lk - 1 : key);
      const __bf16* kb = Km + ((size_t)(b * Lk + keyc)) * D_ + h * DH_;
#pragma unroll
      for (int f = 0; f < 2; ++f) {
        ABu u;
#pragma unroll
        for (int t = 0; t < 8; ++t)
          u.u[t] = *(const unsigned int*)(kb + f * 32 + kpair(t, half));
        s[g] = wmma_bf16(qa[f], u.v, s[g]);
      }
    }

    // scale + ALiBi + mask, online softmax ----------------------------------
#pragma unroll
    for (int v = 0; v < 8; ++v) {
      const int i    = q0 + v + 8 * half;
      float     best = -3.0e38f;
#pragma unroll
      for (int g = 0; g < 2; ++g) {
        const int j  = kc + g * 16 + lr;
        float     sv = s[g][v] * 0.125f + slope * (float)(j - i);
        bool valid = windowed ? (j >= 0 && j <= i && (i - j) < 64) : (j < Lk);
        sv       = valid ? sv : -1.0e30f;
        s[g][v]  = sv;
        best     = fmaxf(best, sv);
      }
#pragma unroll
      for (int mm = 1; mm < 16; mm <<= 1)
        best = fmaxf(best, __shfl_xor(best, mm, 16));
      const float mnew  = fmaxf(mrow[v], best);
      const float alpha = expf(mrow[v] - mnew);
      float psum = 0.0f;
#pragma unroll
      for (int g = 0; g < 2; ++g) {
        float pv = expf(s[g][v] - mnew);
        lP[wave][v + 8 * half][g * 16 + lr] = (__bf16)pv;
        psum += pv;
      }
#pragma unroll
      for (int mm = 1; mm < 16; mm <<= 1) psum += __shfl_xor(psum, mm, 16);
      lrow[v] = lrow[v] * alpha + psum;
      mrow[v] = mnew;
#pragma unroll
      for (int f = 0; f < 4; ++f) o[f][v] *= alpha;
    }
    __syncthreads();

    // P fragment (A operand): 16 queries x 32 keys --------------------------
    v16bf pa;
    {
      ABu u;
#pragma unroll
      for (int t = 0; t < 8; ++t)
        u.u[t] = *(const unsigned int*)&lP[wave][lr][kpair(t, half)];
      pa = u.v;
    }

    // O += P @ V(chunk) ------------------------------------------------------
#pragma unroll
    for (int f = 0; f < 4; ++f) {
      ABu u;
#pragma unroll
      for (int t = 0; t < 8; ++t) {
        int k  = kpair(t, half);
        int j0 = kc + k, j1 = kc + k + 1;
        j0 = j0 < 0 ? 0 : (j0 >= Lk ? Lk - 1 : j0);
        j1 = j1 < 0 ? 0 : (j1 >= Lk ? Lk - 1 : j1);
        u.b[2 * t]     = Vm[((size_t)(b * Lk + j0)) * D_ + h * DH_ + f * 16 + lr];
        u.b[2 * t + 1] = Vm[((size_t)(b * Lk + j1)) * D_ + h * DH_ + f * 16 + lr];
      }
      o[f] = wmma_bf16(pa, u.v, o[f]);
    }
    __syncthreads();
  }

  // normalize + store --------------------------------------------------------
#pragma unroll
  for (int f = 0; f < 4; ++f)
#pragma unroll
    for (int v = 0; v < 8; ++v) {
      const int   row = q0 + v + 8 * half;
      const float inv = 1.0f / fmaxf(lrow[v], 1.0e-20f);
      O[((size_t)(b * Lq + row)) * D_ + h * DH_ + f * 16 + lr] =
          (__bf16)(o[f][v] * inv);
    }
}

// ---------------------------------------------------------------------------
// Fused residual add + LayerNorm over D=1024. One 256-thread block per row.
// Emits f32 (residual stream) and/or bf16 (next-GEMM operand).
// ---------------------------------------------------------------------------
__global__ __launch_bounds__(256) void add_ln_kernel(
    const float* __restrict__ A, const float* __restrict__ Bv,
    const float* __restrict__ g, const float* __restrict__ be,
    float* __restrict__ outF, __bf16* __restrict__ outB) {
  __shared__ float redS[8], redQ[8];
  const int row  = blockIdx.x;
  const int t    = threadIdx.x;
  const int lane = t & 31, wave = t >> 5;
  const float* a = A + (size_t)row * D_;
  const float* r = Bv + (size_t)row * D_;

  float x[4], s = 0.0f, q = 0.0f;
#pragma unroll
  for (int e = 0; e < 4; ++e) {
    int c = t + 256 * e;
    x[e]  = a[c] + r[c];
    s    += x[e];
    q    += x[e] * x[e];
  }
#pragma unroll
  for (int mm = 1; mm < 32; mm <<= 1) {
    s += __shfl_xor(s, mm, 32);
    q += __shfl_xor(q, mm, 32);
  }
  if (lane == 0) { redS[wave] = s; redQ[wave] = q; }
  __syncthreads();
  float S = 0.0f, Q2 = 0.0f;
#pragma unroll
  for (int w = 0; w < 8; ++w) { S += redS[w]; Q2 += redQ[w]; }
  const float mu   = S * (1.0f / D_);
  const float var  = Q2 * (1.0f / D_) - mu * mu;
  const float rstd = rsqrtf(var + 1e-5f);
#pragma unroll
  for (int e = 0; e < 4; ++e) {
    int   c = t + 256 * e;
    float y = (x[e] - mu) * rstd * g[c] + be[c];
    if (outF) outF[(size_t)row * D_ + c] = y;
    if (outB) outB[(size_t)row * D_ + c] = (__bf16)y;
  }
}

// ---------------------------------------------------------------------------
// Host orchestration
// ---------------------------------------------------------------------------
extern "C" void kernel_launch(void* const* d_in, const int* in_sizes, int n_in,
                              void* d_out, int out_size, void* d_ws,
                              size_t ws_size, hipStream_t stream) {
  (void)in_sizes; (void)n_in; (void)out_size; (void)ws_size;
  constexpr int Bc = 2, Lc = 2048, Mc = 2048, Dc = 1024, FFc = 4096;
  const int NT = Bc * Lc;  // 4096 query rows
  const int NM = Bc * Mc;  // 4096 memory rows

  const float* x   = (const float*)d_in[0];
  const float* mem = (const float*)d_in[1];
  const float* swq = (const float*)d_in[2];
  const float* swk = (const float*)d_in[3];
  const float* swv = (const float*)d_in[4];
  const float* swo = (const float*)d_in[5];
  const float* sbq = (const float*)d_in[6];
  const float* sbk = (const float*)d_in[7];
  const float* sbv = (const float*)d_in[8];
  const float* sbo = (const float*)d_in[9];
  const float* cwq = (const float*)d_in[10];
  const float* cwk = (const float*)d_in[11];
  const float* cwv = (const float*)d_in[12];
  const float* cwo = (const float*)d_in[13];
  const float* cbq = (const float*)d_in[14];
  const float* cbk = (const float*)d_in[15];
  const float* cbv = (const float*)d_in[16];
  const float* cbo = (const float*)d_in[17];
  const float* w1  = (const float*)d_in[18];
  const float* b1  = (const float*)d_in[19];
  const float* w2  = (const float*)d_in[20];
  const float* b2  = (const float*)d_in[21];
  const float* g1  = (const float*)d_in[22];
  const float* g2  = (const float*)d_in[23];
  const float* g3  = (const float*)d_in[24];
  const float* be1 = (const float*)d_in[25];
  const float* be2 = (const float*)d_in[26];
  const float* be3 = (const float*)d_in[27];

  char* p = (char*)d_ws;
  auto take = [&](size_t bytes) -> void* {
    void* r = (void*)p;
    p += (bytes + 255) & ~(size_t)255;
    return r;
  };
  __bf16* xcur_bf = (__bf16*)take((size_t)NT * Dc * 2);  // x_bf -> x1_bf -> x2_bf
  __bf16* mem_bf  = (__bf16*)take((size_t)NM * Dc * 2);
  __bf16* swq_bf  = (__bf16*)take((size_t)Dc * Dc * 2);
  __bf16* swk_bf  = (__bf16*)take((size_t)Dc * Dc * 2);
  __bf16* swv_bf  = (__bf16*)take((size_t)Dc * Dc * 2);
  __bf16* swo_bf  = (__bf16*)take((size_t)Dc * Dc * 2);
  __bf16* cwq_bf  = (__bf16*)take((size_t)Dc * Dc * 2);
  __bf16* cwk_bf  = (__bf16*)take((size_t)Dc * Dc * 2);
  __bf16* cwv_bf  = (__bf16*)take((size_t)Dc * Dc * 2);
  __bf16* cwo_bf  = (__bf16*)take((size_t)Dc * Dc * 2);
  __bf16* w1_bf   = (__bf16*)take((size_t)Dc * FFc * 2);
  __bf16* w2_bf   = (__bf16*)take((size_t)FFc * Dc * 2);
  __bf16* q_bf    = (__bf16*)take((size_t)NT * Dc * 2);
  __bf16* k_bf    = (__bf16*)take((size_t)NM * Dc * 2);
  __bf16* v_bf    = (__bf16*)take((size_t)NM * Dc * 2);
  __bf16* attn_bf = (__bf16*)take((size_t)NT * Dc * 2);
  float*  gout    = (float*)take((size_t)NT * Dc * 4);   // sa / ca / ff
  float*  x1f     = (float*)take((size_t)NT * Dc * 4);
  float*  x2f     = (float*)take((size_t)NT * Dc * 4);
  __bf16* h_bf    = (__bf16*)take((size_t)NT * FFc * 2);

  auto cvt = [&](const float* src, __bf16* dst, size_t n) {
    cvt_kernel<<<1024, 256, 0, stream>>>(src, dst, (unsigned long long)n);
  };
  auto gemm = [&](const __bf16* Ag, const __bf16* Wg, const float* biasg,
                  float* Cfg, __bf16* Cbg, int Mg, int Ng, int Kg, int act) {
    dim3 grid(Ng / 128, Mg / 256);
    gemm_bf16_kernel<<<grid, 256, 0, stream>>>(Ag, Wg, biasg, Cfg, Cbg, Mg, Ng,
                                               Kg, act);
  };
  auto attn = [&](const __bf16* Qg, const __bf16* Kg, const __bf16* Vg,
                  __bf16* Og, int Lqv, int Lkv, int windowed) {
    int jobs = Bc * H_ * (Lqv / 16);  // one wave per job, 4 waves per block
    attn_kernel<<<jobs / 4, 128, 0, stream>>>(Qg, Kg, Vg, Og, Lqv, Lkv,
                                              windowed);
  };

  // ---- bf16 staging --------------------------------------------------------
  cvt(x, xcur_bf, (size_t)NT * Dc);
  cvt(mem, mem_bf, (size_t)NM * Dc);
  cvt(swq, swq_bf, (size_t)Dc * Dc);
  cvt(swk, swk_bf, (size_t)Dc * Dc);
  cvt(swv, swv_bf, (size_t)Dc * Dc);
  cvt(swo, swo_bf, (size_t)Dc * Dc);
  cvt(cwq, cwq_bf, (size_t)Dc * Dc);
  cvt(cwk, cwk_bf, (size_t)Dc * Dc);
  cvt(cwv, cwv_bf, (size_t)Dc * Dc);
  cvt(cwo, cwo_bf, (size_t)Dc * Dc);
  cvt(w1, w1_bf, (size_t)Dc * FFc);
  cvt(w2, w2_bf, (size_t)FFc * Dc);

  // ---- self-attention (windowed causal + ALiBi) ----------------------------
  gemm(xcur_bf, swq_bf, sbq, nullptr, q_bf, NT, Dc, Dc, 0);
  gemm(xcur_bf, swk_bf, sbk, nullptr, k_bf, NT, Dc, Dc, 0);
  gemm(xcur_bf, swv_bf, sbv, nullptr, v_bf, NT, Dc, Dc, 0);
  attn(q_bf, k_bf, v_bf, attn_bf, Lc, Lc, 1);
  gemm(attn_bf, swo_bf, sbo, gout, nullptr, NT, Dc, Dc, 0);
  add_ln_kernel<<<NT, 256, 0, stream>>>(x, gout, g1, be1, x1f, xcur_bf);

  // ---- cross-attention (full) ---------------------------------------------
  gemm(xcur_bf, cwq_bf, cbq, nullptr, q_bf, NT, Dc, Dc, 0);
  gemm(mem_bf, cwk_bf, cbk, nullptr, k_bf, NM, Dc, Dc, 0);
  gemm(mem_bf, cwv_bf, cbv, nullptr, v_bf, NM, Dc, Dc, 0);
  attn(q_bf, k_bf, v_bf, attn_bf, Lc, Mc, 0);
  gemm(attn_bf, cwo_bf, cbo, gout, nullptr, NT, Dc, Dc, 0);
  add_ln_kernel<<<NT, 256, 0, stream>>>(x1f, gout, g2, be2, x2f, xcur_bf);

  // ---- FFN -----------------------------------------------------------------
  gemm(xcur_bf, w1_bf, b1, nullptr, h_bf, NT, FFc, Dc, 1);  // fused exact GELU
  gemm(h_bf, w2_bf, b2, gout, nullptr, NT, Dc, FFc, 0);
  add_ln_kernel<<<NT, 256, 0, stream>>>(x2f, gout, g3, be3, (float*)d_out,
                                        nullptr);
}
